// VQVae_68358699483650
// MI455X (gfx1250) — matrix operand order repaired and checked by reading, not compiled, not run on hardware
//
#include <hip/hip_runtime.h>
#include <hip/hip_bf16.h>

// ---------------------------------------------------------------------------
// VQ-VAE forward for MI455X (gfx1250). Conv layers are im2col GEMMs executed
// on the WMMA pipe: v_wmma_f32_16x16x32_bf16, bf16 inputs, f32 accumulation.
// Weights are pre-repacked per layer into the exact A-fragment lane layout so
// the hot loop is: 2x global b128 (A) + 4x ds b128 (B) + 2x WMMA per chunk.
// ---------------------------------------------------------------------------

typedef __attribute__((ext_vector_type(16))) __bf16 v16bf;
typedef __attribute__((ext_vector_type(8)))  __bf16 v8bf;
typedef __attribute__((ext_vector_type(8)))  float  v8f;

#define B_SZ   16
#define T_SZ   2048
#define NFEATS 64
#define WIDTH  512
#define CODES  512

#define BLK_CO 64        // block cout tile (4 wave co-groups of 16)
#define BLK_T  64        // block time tile (2 wave t-groups of 32)
#define XS_PITCH 40      // bf16 elements per window row (32 used, 80B = 16B-aligned)
#define XS_MAX_WIN 132   // max (BLK_T-1)*stride + (K-1)*dil + 1  (=130 for k4 s2)

__device__ __forceinline__ __bf16 to_bf16(float f) {
    union { float f; unsigned u; } uf; uf.f = f;
    unsigned u = uf.u;
    u += 0x7FFFu + ((u >> 16) & 1u);           // round-to-nearest-even
    union { unsigned short s; __bf16 b; } ub;
    ub.s = (unsigned short)(u >> 16);
    return ub.b;
}

#define SHUF16(lo, hi) __builtin_shufflevector(lo, hi, 0,1,2,3,4,5,6,7,8,9,10,11,12,13,14,15)

// ---------------------------------------------------------------------------
// Repack conv weights W:(Cout,Cin,K) f32 -> A-fragment layout bf16:
//   A[((chunk*K + k)*nCoT + coT)*512 + lane*16 + e]
// where lane m-row = coT*16 + (lane&15), ci = chunk*32 + (lane>=16?16:0) + e
// ---------------------------------------------------------------------------
__global__ void repack_w_kernel(const float* __restrict__ W, __bf16* __restrict__ A,
                                int Cout, int Cin, int K)
{
    int idx = blockIdx.x * blockDim.x + threadIdx.x;
    int total = Cout * Cin * K;
    if (idx >= total) return;
    int e    = idx & 15;
    int lane = (idx >> 4) & 31;
    int r    = idx >> 9;
    int nCoT = Cout >> 4;
    int coT  = r % nCoT;
    int r2   = r / nCoT;
    int k    = r2 % K;
    int chunk = r2 / K;
    int m  = coT * 16 + (lane & 15);
    int ci = chunk * 32 + ((lane >= 16) ? 16 : 0) + e;
    A[idx] = to_bf16(W[(size_t)m * Cin * K + (size_t)ci * K + k]);
}

// ---------------------------------------------------------------------------
// Conv1D as GEMM. X:(B,Cin,Tin)  Y:(B,Cout,Tout). 256 threads = 8 waves,
// block tile 64 cout x 64 t, wave tile 16 cout x 32 t (2 accumulators).
// Stage one X window (all K taps) per 32-channel chunk, t-major in LDS.
// ---------------------------------------------------------------------------
__global__ __launch_bounds__(256)
void conv1d_wmma_kernel(const float* __restrict__ X,
                        const __bf16* __restrict__ Arep,
                        const float* __restrict__ Bias,
                        const float* Res,          // optional residual (may alias Y)
                        float* Y,
                        int Cin, int Cout, int Tin, int Tout,
                        int K, int stride, int pad, int dil,
                        int reluIn, int reluOut)
{
    __shared__ __bf16 Xs[XS_MAX_WIN * XS_PITCH];

    const int tid    = threadIdx.x;
    const int lane   = tid & 31;
    const int wave   = tid >> 5;
    const int coBase = blockIdx.x * BLK_CO;
    const int tBase  = blockIdx.y * BLK_T;
    const int bb     = blockIdx.z;
    const int waveCo = (wave >> 1) * 16;   // 4 cout sub-tiles
    const int waveT  = (wave & 1) * 32;    // 2 time groups of 32 (2 accums each)
    const int nCoT   = Cout >> 4;
    const int coTile = (coBase + waveCo) >> 4;

    const int Wwin = (BLK_T - 1) * stride + (K - 1) * dil + 1;
    const int orig = tBase * stride - pad;      // window origin in input

    const int kbBase = (lane >= 16) ? 16 : 0;   // B-fragment K-half select
    const int n0 = waveT + (lane & 15);         // t within block tile, accum 0
    const int n1 = n0 + 16;                     // accum 1

    v8f acc0 = {}, acc1 = {};
    const size_t xbase = (size_t)bb * Cin * Tin;

    for (int cb = 0; cb < Cin; cb += 32) {
        __syncthreads();
        // stage X window: rows = window pos (t-major), cols = 32 channels.
        // global reads coalesced along t; clamp+select avoids divergent loads.
        for (int ci = wave; ci < 32; ci += 8) {
            const float* xrow = X + xbase + (size_t)(cb + ci) * Tin;
            for (int wp = lane; wp < Wwin; wp += 32) {
                int it  = orig + wp;
                int itc = min(max(it, 0), Tin - 1);
                float v = xrow[itc];
                v = (it >= 0 && it < Tin) ? v : 0.f;
                if (reluIn) v = fmaxf(v, 0.f);
                Xs[wp * XS_PITCH + ci] = to_bf16(v);
            }
        }
        __syncthreads();

        for (int k = 0; k < K; ++k) {
            // A fragment: pre-repacked, 32 contiguous bytes per lane.
            const __bf16* ap =
                Arep + ((((size_t)(cb >> 5)) * K + k) * nCoT + coTile) * 512 + lane * 16;
            v8bf alo = *(const v8bf*)ap;
            v8bf ahi = *(const v8bf*)(ap + 8);
            v16bf a  = SHUF16(alo, ahi);

            // B fragments: two ds_load_b128 each (16B-aligned rows).
            const __bf16* bp0 = &Xs[(n0 * stride + k * dil) * XS_PITCH + kbBase];
            v8bf b0lo = *(const v8bf*)bp0;
            v8bf b0hi = *(const v8bf*)(bp0 + 8);
            v16bf bm0 = SHUF16(b0lo, b0hi);
            acc0 = __builtin_amdgcn_wmma_f32_16x16x32_bf16(
                       false, a, false, bm0, (short)0, acc0, false, false);

            const __bf16* bp1 = &Xs[(n1 * stride + k * dil) * XS_PITCH + kbBase];
            v8bf b1lo = *(const v8bf*)bp1;
            v8bf b1hi = *(const v8bf*)(bp1 + 8);
            v16bf bm1 = SHUF16(b1lo, b1hi);
            acc1 = __builtin_amdgcn_wmma_f32_16x16x32_bf16(
                       false, a, false, bm1, (short)0, acc1, false, false);
        }
    }

    // epilogue: bias + optional residual + optional relu
    const int coHalf = (lane >= 16) ? 8 : 0;
    const int tg0 = tBase + n0;
    const int tg1 = tBase + n1;
#pragma unroll
    for (int v = 0; v < 8; ++v) {
        int co = coBase + waveCo + v + coHalf;
        if (co < Cout) {
            float bv = Bias[co];
            size_t orow = ((size_t)bb * Cout + co) * Tout;
            if (tg0 < Tout) {
                float y = acc0[v] + bv;
                if (Res) y += Res[orow + tg0];
                if (reluOut) y = fmaxf(y, 0.f);
                Y[orow + tg0] = y;
            }
            if (tg1 < Tout) {
                float y = acc1[v] + bv;
                if (Res) y += Res[orow + tg1];
                if (reluOut) y = fmaxf(y, 0.f);
                Y[orow + tg1] = y;
            }
        }
    }
}

// (B,T,C) -> (B,C,T)
__global__ void transpose_in_kernel(const float* __restrict__ F, float* __restrict__ X, size_t n) {
    size_t i = (size_t)blockIdx.x * blockDim.x + threadIdx.x;
    if (i >= n) return;
    size_t b = i / ((size_t)T_SZ * NFEATS);
    size_t r = i % ((size_t)T_SZ * NFEATS);
    size_t c = r / T_SZ, t = r % T_SZ;
    X[i] = F[b * T_SZ * NFEATS + t * NFEATS + c];
}

// (B,C,T) -> (B,T,C)
__global__ void transpose_out_kernel(const float* __restrict__ X, float* __restrict__ O, size_t n) {
    size_t i = (size_t)blockIdx.x * blockDim.x + threadIdx.x;
    if (i >= n) return;
    size_t b = i / ((size_t)T_SZ * NFEATS);
    size_t r = i % ((size_t)T_SZ * NFEATS);
    size_t t = r / NFEATS, c = r % NFEATS;
    O[i] = X[(b * NFEATS + c) * T_SZ + t];
}

__global__ void upsample2_kernel(const float* __restrict__ X, float* __restrict__ Y,
                                 size_t n, int Tin) {
    size_t i = (size_t)blockIdx.x * blockDim.x + threadIdx.x;
    if (i >= n) return;
    size_t row = i / Tin;
    int t = (int)(i % Tin);
    float v = X[i];
    Y[row * (size_t)(2 * Tin) + 2 * t]     = v;
    Y[row * (size_t)(2 * Tin) + 2 * t + 1] = v;
}

__global__ void zero_kernel(float* p, int n) {
    int i = blockIdx.x * blockDim.x + threadIdx.x;
    if (i < n) p[i] = 0.f;
}

// One wave per token: argmin over 512 codes, write quantized row, accumulate
// commit-loss and code histogram.
__global__ __launch_bounds__(256)
void quantize_kernel(const float* __restrict__ XENC, const float* __restrict__ CB,
                     float* __restrict__ XQ, float* counts, float* commitAcc,
                     int C, int Tq, int nTok)
{
    int w    = (int)(((size_t)blockIdx.x * blockDim.x + threadIdx.x) >> 5);
    int lane = threadIdx.x & 31;
    if (w >= nTok) return;
    int b = w / Tq, t = w % Tq;
    const float* x = XENC + (size_t)b * C * Tq + t;   // stride Tq per channel

    float best = 3.4e38f; int bestj = 0;
    for (int j = 0; j < CODES; ++j) {
        const float* c = CB + (size_t)j * C;
        float dot = 0.f, cc = 0.f;
        for (int d = lane; d < C; d += 32) {
            float cv = c[d];
            dot += x[(size_t)d * Tq] * cv;
            cc  += cv * cv;
        }
        for (int off = 16; off; off >>= 1) {
            dot += __shfl_xor(dot, off, 32);
            cc  += __shfl_xor(cc,  off, 32);
        }
        float dist = cc - 2.f * dot;                  // ||x||^2 constant for argmin
        if (dist < best) { best = dist; bestj = j; }
    }
    const float* c = CB + (size_t)bestj * C;
    float se = 0.f;
    for (int d = lane; d < C; d += 32) {
        float cv = c[d];
        float xv = x[(size_t)d * Tq];
        XQ[(size_t)b * C * Tq + (size_t)d * Tq + t] = cv;
        float df = xv - cv; se += df * df;
    }
    for (int off = 16; off; off >>= 1) se += __shfl_xor(se, off, 32);
    if (lane == 0) {
        atomicAdd(commitAcc, se);
        atomicAdd(&counts[bestj], 1.0f);
    }
}

// Rotation trick along time axis, one wave per (b,c) row of length Trow.
// R@x = x - 2 r (r.x) + 2 q (e.x); closed forms from Sxx, Sqq, Sxq.
__global__ __launch_bounds__(256)
void rotation_kernel(const float* __restrict__ XENC, float* XQ, int Trow, int nRows)
{
    int w    = (int)(((size_t)blockIdx.x * blockDim.x + threadIdx.x) >> 5);
    int lane = threadIdx.x & 31;
    if (w >= nRows) return;
    const float* x = XENC + (size_t)w * Trow;
    float* q = XQ + (size_t)w * Trow;

    float sxx = 0.f, sqq = 0.f, sxq = 0.f;
    for (int t = lane; t < Trow; t += 32) {
        float xv = x[t], qv = q[t];
        sxx += xv * xv; sqq += qv * qv; sxq += xv * qv;
    }
    for (int off = 16; off; off >>= 1) {
        sxx += __shfl_xor(sxx, off, 32);
        sqq += __shfl_xor(sqq, off, 32);
        sxq += __shfl_xor(sxq, off, 32);
    }
    float ne = fmaxf(sqrtf(fmaxf(sxx, 0.f)), 1e-12f);
    float nq = fmaxf(sqrtf(fmaxf(sqq, 0.f)), 1e-12f);
    float eq = sxq / (ne * nq);
    float nr = fmaxf(sqrtf(fmaxf(2.f + 2.f * eq, 0.f)), 1e-12f);
    float rx  = (ne + sxq / nq) / nr;   // sum r_t * x_t
    float ex  = ne;                     // sum e_t * x_t
    float scl = nq / ne;
    for (int t = lane; t < Trow; t += 32) {
        float xv = x[t], qv = q[t];
        float rt = (xv / ne + qv / nq) / nr;
        float Rx = xv - 2.f * rt * rx + 2.f * (qv / nq) * ex;
        q[t] = scl * Rx;
    }
}

__global__ void finalize_kernel(const float* counts, const float* commitAcc,
                                float* outScal, int nTok, int C)
{
    __shared__ float sh[CODES];
    int tid = threadIdx.x;
    float p = counts[tid] / (float)nTok;
    sh[tid] = p * logf(p + 1e-7f);
    __syncthreads();
    for (int s = CODES / 2; s > 0; s >>= 1) {
        if (tid < s) sh[tid] += sh[tid + s];
        __syncthreads();
    }
    if (tid == 0) {
        outScal[0] = commitAcc[0] / ((float)nTok * (float)C);  // commit
        outScal[1] = expf(-sh[0]);                             // perplexity
    }
}

// ---------------------------------------------------------------------------
// Host-side orchestration
// ---------------------------------------------------------------------------
struct ConvP { const float* w; const float* b; };

struct PCursor {
    void* const* din; int idx;          // per-leaf mode
    const float* flat; size_t off;      // flat mode
    bool perLeaf;
    const float* take(size_t n) {
        if (perLeaf) return (const float*)din[idx++];
        const float* r = flat + off; off += n; return r;
    }
    ConvP conv(int O, int I, int K) {
        ConvP p;
        p.b = take((size_t)O);                  // 'b' < 'w' in jax sorted-key order
        p.w = take((size_t)O * I * K);
        return p;
    }
};

extern "C" void kernel_launch(void* const* d_in, const int* in_sizes, int n_in,
                              void* d_out, int out_size, void* d_ws, size_t ws_size,
                              hipStream_t stream)
{
    const float* features = (const float*)d_in[0];

    PCursor pc;
    if (n_in >= 96) {                       // one pointer per param leaf
        pc.din = d_in; pc.idx = 1; pc.flat = nullptr; pc.off = 0; pc.perLeaf = true;
    } else {                                // params passed as one flat buffer
        pc.din = d_in; pc.idx = 0; pc.flat = (const float*)d_in[1]; pc.off = 0; pc.perLeaf = false;
    }

    // jax tree order: dict keys sorted at every level ('dec' < 'enc', 'b' < 'w')
    ConvP dec_conv_in  = pc.conv(WIDTH, WIDTH, 3);
    ConvP dec_conv_mid = pc.conv(WIDTH, WIDTH, 3);
    ConvP dec_conv_out = pc.conv(NFEATS, WIDTH, 3);
    ConvP dec_ups_conv[3], dec_c1[3][3], dec_c2[3][3];
    for (int i = 0; i < 3; ++i) {
        dec_ups_conv[i] = pc.conv(WIDTH, WIDTH, 3);  // 'conv' < 'res'
        for (int j = 0; j < 3; ++j) {
            dec_c1[i][j] = pc.conv(WIDTH, WIDTH, 3);
            dec_c2[i][j] = pc.conv(WIDTH, WIDTH, 1);
        }
    }
    ConvP enc_conv_in  = pc.conv(WIDTH, NFEATS, 3);
    ConvP enc_conv_out = pc.conv(WIDTH, WIDTH, 3);
    ConvP enc_down[3], enc_c1[3][3], enc_c2[3][3];
    for (int i = 0; i < 3; ++i) {
        enc_down[i] = pc.conv(WIDTH, WIDTH, 4);      // 'down' < 'res'
        for (int j = 0; j < 3; ++j) {
            enc_c1[i][j] = pc.conv(WIDTH, WIDTH, 3);
            enc_c2[i][j] = pc.conv(WIDTH, WIDTH, 1);
        }
    }
    const float* codebook = pc.perLeaf ? (const float*)d_in[n_in - 1]
                                       : (const float*)d_in[2];

    // workspace layout (floats, all offsets 32B-aligned)
    float* ws = (float*)d_ws;
    const size_t SZ_SMALL = (size_t)B_SZ * NFEATS * T_SZ;        //  2,097,152
    const size_t SZ_BIG   = (size_t)B_SZ * WIDTH  * T_SZ;        // 16,777,216
    const size_t SZ_ENC   = (size_t)B_SZ * WIDTH  * (T_SZ / 8);  //  2,097,152
    const size_t SZ_RPK_F = ((size_t)WIDTH * WIDTH * 4) / 2;     // 1M bf16 = 512K floats
    float*  XIN    = ws;                      // reused for decoder pre-transpose out
    float*  BUFA   = XIN  + SZ_SMALL;
    float*  BUFB   = BUFA + SZ_BIG;
    float*  XENC   = BUFB + SZ_BIG;
    float*  XQ     = XENC + SZ_ENC;
    __bf16* RPK    = (__bf16*)(XQ + SZ_ENC);  // per-layer repacked weights
    float*  COUNTS = XQ + SZ_ENC + SZ_RPK_F;
    float*  COMMIT = COUNTS + CODES;

    auto conv = [&](const float* X, ConvP p, const float* res, float* Y,
                    int Cin, int Cout, int Tin, int Tout,
                    int K, int s, int pad, int dil, int ri, int ro) {
        int total = Cout * Cin * K;
        repack_w_kernel<<<(total + 255) / 256, 256, 0, stream>>>(p.w, RPK, Cout, Cin, K);
        dim3 g((Cout + BLK_CO - 1) / BLK_CO, (Tout + BLK_T - 1) / BLK_T, B_SZ);
        conv1d_wmma_kernel<<<g, 256, 0, stream>>>(X, RPK, p.b, res, Y,
                                                  Cin, Cout, Tin, Tout,
                                                  K, s, pad, dil, ri, ro);
    };

    // re-zero accumulators every call (deterministic replay)
    zero_kernel<<<(CODES + 8 + 255) / 256, 256, 0, stream>>>(COUNTS, CODES + 8);

    // ---------------- encoder ----------------
    transpose_in_kernel<<<(unsigned)((SZ_SMALL + 255) / 256), 256, 0, stream>>>(features, XIN, SZ_SMALL);
    conv(XIN, enc_conv_in, nullptr, BUFA, NFEATS, WIDTH, T_SZ, T_SZ, 3, 1, 1, 1, 0, 1);

    float* cur = BUFA; float* oth = BUFB;
    int T = T_SZ;
    for (int i = 0; i < 3; ++i) {
        conv(cur, enc_down[i], nullptr, oth, WIDTH, WIDTH, T, T / 2, 4, 2, 1, 1, 0, 0);
        { float* tmp = cur; cur = oth; oth = tmp; }
        T /= 2;
        int dil = 1;
        for (int j = 0; j < 3; ++j) {
            conv(cur, enc_c1[i][j], nullptr, oth, WIDTH, WIDTH, T, T, 3, 1, dil, dil, 1, 0);
            conv(oth, enc_c2[i][j], cur,     cur, WIDTH, WIDTH, T, T, 1, 1, 0,   1,   1, 0);
            dil *= 3;
        }
    }
    conv(cur, enc_conv_out, nullptr, XENC, WIDTH, WIDTH, T, T, 3, 1, 1, 1, 0, 0);

    // ---------------- quantize + rotation ----------------
    const int Tq = T_SZ / 8, nTok = B_SZ * Tq;
    quantize_kernel<<<(nTok * 32 + 255) / 256, 256, 0, stream>>>(
        XENC, codebook, XQ, COUNTS, COMMIT, WIDTH, Tq, nTok);
    const int nRows = B_SZ * WIDTH;
    rotation_kernel<<<(nRows * 32 + 255) / 256, 256, 0, stream>>>(XENC, XQ, Tq, nRows);

    // ---------------- decoder ----------------
    conv(XQ, dec_conv_in, nullptr, BUFA, WIDTH, WIDTH, Tq, Tq, 3, 1, 1, 1, 0, 1);
    cur = BUFA; oth = BUFB; T = Tq;
    for (int i = 0; i < 3; ++i) {
        int dil = 9;
        for (int j = 0; j < 3; ++j) {
            conv(cur, dec_c1[i][j], nullptr, oth, WIDTH, WIDTH, T, T, 3, 1, dil, dil, 1, 0);
            conv(oth, dec_c2[i][j], cur,     cur, WIDTH, WIDTH, T, T, 1, 1, 0,   1,   1, 0);
            dil /= 3;
        }
        size_t nUp = (size_t)B_SZ * WIDTH * T;
        upsample2_kernel<<<(unsigned)((nUp + 255) / 256), 256, 0, stream>>>(cur, oth, nUp, T);
        T *= 2;
        { float* tmp = cur; cur = oth; oth = tmp; }
        conv(cur, dec_ups_conv[i], nullptr, oth, WIDTH, WIDTH, T, T, 3, 1, 1, 1, 0, 0);
        { float* tmp = cur; cur = oth; oth = tmp; }
    }
    conv(cur, dec_conv_mid, nullptr, oth, WIDTH, WIDTH, T_SZ, T_SZ, 3, 1, 1, 1, 0, 1);
    { float* tmp = cur; cur = oth; oth = tmp; }
    conv(cur, dec_conv_out, nullptr, XIN, WIDTH, NFEATS, T_SZ, T_SZ, 3, 1, 1, 1, 0, 0);

    // ---------------- outputs ----------------
    float* out = (float*)d_out;
    transpose_out_kernel<<<(unsigned)((SZ_SMALL + 255) / 256), 256, 0, stream>>>(XIN, out, SZ_SMALL);
    finalize_kernel<<<1, CODES, 0, stream>>>(COUNTS, COMMIT, out + SZ_SMALL, nTok, WIDTH);
}
